// SceneGraphEmbedding_65652870086842
// MI455X (gfx1250) — matrix operand order repaired and checked by reading, not compiled, not run on hardware
//
#include <hip/hip_runtime.h>

typedef __attribute__((ext_vector_type(16))) _Float16 v16h;
typedef __attribute__((ext_vector_type(8)))  _Float16 v8h;
typedef __attribute__((ext_vector_type(8)))  float    v8f;

// ---------------- problem constants ----------------
#define NN    20000
#define EE    320000
#define E2E   340000     // E + N (self loops)
#define GGR   256
#define HCN   32
#define HEN   32
#define HIDN  64
#define H1N   4
#define HO1N  256        // H1N * HIDN
#define OUTN  128

static inline int cdiv_h(long a, long b) { return (int)((a + b - 1) / b); }

// ---------------- monotonic float<->uint key (for atomic max on f32) ------
__device__ __forceinline__ unsigned fkey_enc(float f) {
    unsigned u = __float_as_uint(f);
    return (u & 0x80000000u) ? ~u : (u | 0x80000000u);
}
__device__ __forceinline__ float fkey_dec(unsigned k) {
    unsigned u = (k & 0x80000000u) ? (k ^ 0x80000000u) : ~k;
    return __uint_as_float(u);
}

// ---------------- generic zero ----------------
__global__ void k_zero(float* __restrict__ p, long n) {
    long i = (long)blockIdx.x * blockDim.x + threadIdx.x;
    if (i < n) p[i] = 0.0f;
}

// ---------------- f32 -> f16 convert ----------------
__global__ void k_cvt(const float* __restrict__ in, _Float16* __restrict__ out, long n) {
    long i = (long)blockIdx.x * blockDim.x + threadIdx.x;
    if (i < n) out[i] = (_Float16)in[i];
}

// ---------------- transpose + convert: in f32 [R,C] -> out f16 [C,R] ------
__global__ void k_transpose_h(const float* __restrict__ in, _Float16* __restrict__ out,
                              int R, int C) {
    int i = blockIdx.x * blockDim.x + threadIdx.x;
    if (i >= R * C) return;
    int r = i / C, c = i - r * C;
    out[(long)c * R + r] = (_Float16)in[i];
}

// ---------------- node init: s=relu(x[:,:16]@Wc+bc); h0=relu([x[:,16:24],s]@Wi+bi)
__global__ __launch_bounds__(64)
void k_node_init(const float* __restrict__ x, const float* __restrict__ Wc,
                 const float* __restrict__ bc, const float* __restrict__ Wi,
                 const float* __restrict__ bi, _Float16* __restrict__ h0h) {
    __shared__ float s[HCN];
    int n = blockIdx.x;
    int t = threadIdx.x;
    const float* xr = x + (long)n * 24;
    if (t < HCN) {
        float acc = bc[t];
        #pragma unroll
        for (int k = 0; k < 16; ++k) acc += xr[k] * Wc[k * HCN + t];
        s[t] = fmaxf(acc, 0.0f);
    }
    __syncthreads();
    float acc = bi[t];
    #pragma unroll
    for (int k = 0; k < 8; ++k) acc += xr[16 + k] * Wi[k * HIDN + t];
    #pragma unroll
    for (int k = 0; k < HCN; ++k) acc += s[k] * Wi[(8 + k) * HIDN + t];
    h0h[(long)n * HIDN + t] = (_Float16)fmaxf(acc, 0.0f);
}

// ---------------- edge embedding: ee2[e,:] = edge_attr[e,:9] @ We + be ----
__global__ void k_edge_embed(const float* __restrict__ ea, const float* __restrict__ We,
                             const float* __restrict__ be, float* __restrict__ ee2) {
    long i = (long)blockIdx.x * blockDim.x + threadIdx.x;
    if (i >= (long)EE * HEN) return;
    int e = (int)(i >> 5), j = (int)(i & 31);
    const float* r = ea + (long)e * 9;
    float acc = be[j];
    #pragma unroll
    for (int k = 0; k < 9; ++k) acc += r[k] * We[k * HEN + j];
    ee2[i] = acc;
}

// ---------------- self-loop attr: mean of ee over tgt -> ee2 rows [E,E2) --
__global__ void k_loop_accum(float* __restrict__ ee2, const int* __restrict__ tgt,
                             float* __restrict__ cnt) {
    long i = (long)blockIdx.x * blockDim.x + threadIdx.x;
    if (i >= (long)EE * HEN) return;
    int e = (int)(i >> 5), j = (int)(i & 31);
    int t = tgt[e];
    atomicAdd(&ee2[((long)EE + t) * HEN + j], ee2[i]);
    if (j == 0) atomicAdd(&cnt[t], 1.0f);
}
__global__ void k_loop_final(float* __restrict__ ee2, const float* __restrict__ cnt) {
    long i = (long)blockIdx.x * blockDim.x + threadIdx.x;
    if (i >= (long)NN * HEN) return;
    int n = (int)(i >> 5);
    ee2[(long)EE * HEN + i] /= fmaxf(cnt[n], 1.0f);
}

// ---------------- WMMA GEMM: C[M,Nc] = A[M,K](f16,row) @ Bt[Nc,K]^T + bias -
// One wave owns one 16-row strip: A fragments loaded ONCE into VGPRs, then
// swept across all Nt column tiles (B tiles are tiny & cache-resident).
// Cuts redundant A traffic by Nt x vs tile-per-wave (e.g. 330MB on ep1 GEMM).
template <int K>
__global__ __launch_bounds__(128)
void k_gemm_wmma(const _Float16* __restrict__ A, const _Float16* __restrict__ Bt,
                 const float* __restrict__ bias, float* __restrict__ C,
                 int M, int Nc) {
    constexpr int KS = K / 32;
    int strip = blockIdx.x * (blockDim.x >> 5) + (threadIdx.x >> 5);
    int Mt = M >> 4;
    if (strip >= Mt) return;                      // wave-uniform guard
    int Nt = Nc >> 4;
    int lane = threadIdx.x & 31;
    int r = lane & 15, g = lane >> 4;

    // load all A fragments for this strip once (held in registers)
    const _Float16* arow = A + (long)(strip * 16 + r) * K;
    v16h a[KS];
    #pragma unroll
    for (int ks = 0; ks < KS; ++ks) {
        // A lane layout: halves 0..15 = K[ks*32+g*16 .. +15], contiguous
        v8h lo = *(const v8h*)(arow + ks * 32 + g * 16);
        v8h hi = *(const v8h*)(arow + ks * 32 + g * 16 + 8);
        #pragma unroll
        for (int i = 0; i < 8; ++i) { a[ks][i] = lo[i]; a[ks][i + 8] = hi[i]; }
    }

    const _Float16* bcol = Bt + (long)r * K + (long)g * 16;
    for (int nt = 0; nt < Nt; ++nt) {
        if (nt + 1 < Nt)                          // hint next B tile into cache
            __builtin_prefetch(bcol + (long)(nt + 1) * 16 * K, 0, 0);
        v8f c = {};
        #pragma unroll
        for (int ks = 0; ks < KS; ++ks) {
            // B lane layout: halves i = K[ks*32 + g*16 + i], contiguous 32B
            v16h b = *(const v16h*)(bcol + (long)nt * 16 * K + ks * 32);
            c = __builtin_amdgcn_wmma_f32_16x16x32_f16(false, a[ks], false, b,
                                                       (short)0, c, false, false);
        }
        int col = nt * 16 + r;
        float bv = bias ? bias[col] : 0.0f;
        #pragma unroll
        for (int j = 0; j < 8; ++j) {
            int row = strip * 16 + j + g * 8;     // C/D layout: VGPR j <-> row j+g*8
            C[(long)row * Nc + col] = c[j] + bv;
        }
    }
}

// ---------------- attention scores: one wave per edge ---------------------
template <int HOUT, int H>
__global__ __launch_bounds__(128)
void k_scores(const float* __restrict__ xl, const float* __restrict__ xr,
              const float* __restrict__ ep, const int* __restrict__ src,
              const int* __restrict__ tgt, int E0, int E2tot,
              const float* __restrict__ att, float* __restrict__ score,
              unsigned* __restrict__ emax) {
    int e = blockIdx.x * 4 + (threadIdx.x >> 5);
    if (e >= E2tot) return;
    int lane = threadIdx.x & 31;
    int s, t;
    if (e < E0) { s = src[e]; t = tgt[e]; } else { s = e - E0; t = s; }
    constexpr int cpl = HOUT / 32;
    int c0 = lane * cpl;
    float part = 0.0f;
    #pragma unroll
    for (int i = 0; i < cpl; ++i) {
        int col = c0 + i;
        float m = xl[(long)s * HOUT + col] + xr[(long)t * HOUT + col]
                + ep[(long)e * HOUT + col];
        m = m > 0.0f ? m : 0.2f * m;              // leaky_relu(0.2)
        part += m * att[col];
    }
    constexpr int gsz = 32 / H;                   // lanes per head
    #pragma unroll
    for (int m = gsz >> 1; m >= 1; m >>= 1) part += __shfl_xor(part, m, 32);
    if ((lane & (gsz - 1)) == 0) {
        int h = lane / gsz;
        score[(long)e * H + h] = part;
        atomicMax(&emax[t * H + h], fkey_enc(part));
    }
}

// ---------------- softmax numerator + denominator -------------------------
__global__ void k_exp(float* __restrict__ score, const unsigned* __restrict__ emax,
                      float* __restrict__ den, const int* __restrict__ tgt,
                      int E0, int E2tot, int H) {
    long i = (long)blockIdx.x * blockDim.x + threadIdx.x;
    if (i >= (long)E2tot * H) return;
    int e = (int)(i / H), h = (int)(i - (long)e * H);
    int t = (e < E0) ? tgt[e] : (e - E0);
    float p = __expf(score[i] - fkey_dec(emax[t * H + h]));
    score[i] = p;
    atomicAdd(&den[t * H + h], p);
}

// ---------------- weighted scatter: out[tgt] += alpha * xl[src] -----------
template <int HOUT, int H>
__global__ __launch_bounds__(128)
void k_scatter(const float* __restrict__ xl, const int* __restrict__ src,
               const int* __restrict__ tgt, int E0, int E2tot,
               const float* __restrict__ score, const float* __restrict__ den,
               float* __restrict__ out) {
    int e = blockIdx.x * 4 + (threadIdx.x >> 5);
    if (e >= E2tot) return;
    int lane = threadIdx.x & 31;
    int s, t;
    if (e < E0) { s = src[e]; t = tgt[e]; } else { s = e - E0; t = s; }
    constexpr int cpl = HOUT / 32;
    int c0 = lane * cpl;
    int h = c0 >> 6;                              // 64 channels per head
    float alpha = score[(long)e * H + h] / (den[t * H + h] + 1e-16f);
    #pragma unroll
    for (int i = 0; i < cpl; ++i) {
        int col = c0 + i;
        atomicAdd(&out[(long)t * HOUT + col], alpha * xl[(long)s * HOUT + col]);
    }
}

// ---------------- bias + elu epilogue (optional f32 / f16 outputs) --------
__global__ void k_epilogue(const float* __restrict__ acc, const float* __restrict__ bias,
                           float* __restrict__ fout, _Float16* __restrict__ hout,
                           int HOUT, long n) {
    long i = (long)blockIdx.x * blockDim.x + threadIdx.x;
    if (i >= n) return;
    int col = (int)(i % HOUT);
    float v = acc[i] + bias[col];
    v = v > 0.0f ? v : (__expf(v) - 1.0f);        // elu
    if (fout) fout[i] = v;
    if (hout) hout[i] = (_Float16)v;
}

// ---------------- global mean pool ----------------------------------------
__global__ void k_pool(const float* __restrict__ h2, const int* __restrict__ batch,
                       float* __restrict__ pooled, float* __restrict__ gcnt) {
    long i = (long)blockIdx.x * blockDim.x + threadIdx.x;
    if (i >= (long)NN * HIDN) return;
    int n = (int)(i >> 6), j = (int)(i & 63);
    int g = batch[n];
    atomicAdd(&pooled[g * HIDN + j], h2[i]);
    if (j == 0) atomicAdd(&gcnt[g], 1.0f);
}
__global__ void k_pool_div(float* __restrict__ pooled, const float* __restrict__ gcnt,
                           _Float16* __restrict__ pooledh) {
    int i = blockIdx.x * blockDim.x + threadIdx.x;
    if (i >= GGR * HIDN) return;
    int g = i >> 6;
    float v = pooled[i] / fmaxf(gcnt[g], 1.0f);
    pooled[i] = v;
    pooledh[i] = (_Float16)v;
}

// ======================================================================
extern "C" void kernel_launch(void* const* d_in, const int* in_sizes, int n_in,
                              void* d_out, int out_size, void* d_ws, size_t ws_size,
                              hipStream_t stream) {
    (void)in_sizes; (void)n_in; (void)out_size; (void)ws_size;
    const float* x     = (const float*)d_in[0];
    const int*   eidx  = (const int*)d_in[1];
    const float* ea    = (const float*)d_in[2];
    const int*   batch = (const int*)d_in[3];
    const float* Wc  = (const float*)d_in[4];  const float* bc  = (const float*)d_in[5];
    const float* We  = (const float*)d_in[6];  const float* be  = (const float*)d_in[7];
    const float* Wi  = (const float*)d_in[8];  const float* bi  = (const float*)d_in[9];
    const float* Wl1 = (const float*)d_in[10]; const float* bl1 = (const float*)d_in[11];
    const float* Wr1 = (const float*)d_in[12]; const float* br1 = (const float*)d_in[13];
    const float* We1 = (const float*)d_in[14];
    const float* att1  = (const float*)d_in[15];
    const float* bias1 = (const float*)d_in[16];
    const float* Wl2 = (const float*)d_in[17]; const float* bl2 = (const float*)d_in[18];
    const float* Wr2 = (const float*)d_in[19]; const float* br2 = (const float*)d_in[20];
    const float* We2 = (const float*)d_in[21];
    const float* att2  = (const float*)d_in[22];
    const float* bias2 = (const float*)d_in[23];
    const float* Wo  = (const float*)d_in[24]; const float* bo  = (const float*)d_in[25];

    const int* srcp = eidx;         // edge_index[0]
    const int* tgtp = eidx + EE;    // edge_index[1]

    // ---- workspace bump allocator (256B aligned) ----
    char* wp = (char*)d_ws;
    auto alloc = [&](size_t bytes) -> void* {
        void* r = (void*)wp;
        wp += (bytes + 255) & ~(size_t)255;
        return r;
    };
    _Float16* h0h   = (_Float16*)alloc((size_t)NN * HIDN * 2);
    float*    ee2   = (float*)alloc((size_t)E2E * HEN * 4);
    _Float16* ee2h  = (_Float16*)alloc((size_t)E2E * HEN * 2);
    float*    cnt   = (float*)alloc((size_t)NN * 4);
    float*    xl1   = (float*)alloc((size_t)NN * HO1N * 4);
    float*    xr1   = (float*)alloc((size_t)NN * HO1N * 4);
    float*    ep    = (float*)alloc((size_t)E2E * HO1N * 4);   // reused layer2 (E2*64)
    float*    sc1   = (float*)alloc((size_t)E2E * H1N * 4);
    unsigned* emax1 = (unsigned*)alloc((size_t)NN * H1N * 4);
    float*    den1  = (float*)alloc((size_t)NN * H1N * 4);
    float*    out1  = (float*)alloc((size_t)NN * HO1N * 4);
    _Float16* h1h   = (_Float16*)alloc((size_t)NN * HO1N * 2);
    float*    xl2   = (float*)alloc((size_t)NN * HIDN * 4);
    float*    xr2   = (float*)alloc((size_t)NN * HIDN * 4);
    float*    sc2   = (float*)alloc((size_t)E2E * 4);
    unsigned* emax2 = (unsigned*)alloc((size_t)NN * 4);
    float*    den2  = (float*)alloc((size_t)NN * 4);
    float*    out2  = (float*)alloc((size_t)NN * HIDN * 4);
    float*    pooled = (float*)alloc((size_t)GGR * HIDN * 4);
    float*    gcnt   = (float*)alloc((size_t)GGR * 4);
    _Float16* pooledh = (_Float16*)alloc((size_t)GGR * HIDN * 2);
    _Float16* wl1t = (_Float16*)alloc(256 * 64 * 2);
    _Float16* wr1t = (_Float16*)alloc(256 * 64 * 2);
    _Float16* we1t = (_Float16*)alloc(256 * 32 * 2);
    _Float16* wl2t = (_Float16*)alloc(64 * 256 * 2);
    _Float16* wr2t = (_Float16*)alloc(64 * 256 * 2);
    _Float16* we2t = (_Float16*)alloc(64 * 32 * 2);
    _Float16* wot  = (_Float16*)alloc(128 * 64 * 2);

    auto zero = [&](void* p, long n) {
        k_zero<<<cdiv_h(n, 256), 256, 0, stream>>>((float*)p, n);
    };
    auto gemm = [&](const _Float16* A, const _Float16* Bt, const float* bias,
                    float* C, int M, int Nc, int K) {
        int blocks = cdiv_h(M / 16, 4);           // 4 strips (waves) per block
        if (K == 32)
            k_gemm_wmma<32><<<blocks, 128, 0, stream>>>(A, Bt, bias, C, M, Nc);
        else if (K == 64)
            k_gemm_wmma<64><<<blocks, 128, 0, stream>>>(A, Bt, bias, C, M, Nc);
        else
            k_gemm_wmma<256><<<blocks, 128, 0, stream>>>(A, Bt, bias, C, M, Nc);
    };

    // ---- zero all accumulators (ws is poisoned before timing) ----
    zero(ee2 + (long)EE * HEN, (long)NN * HEN);   // self-loop attr rows
    zero(cnt, NN);
    zero(emax1, (long)NN * H1N); zero(den1, (long)NN * H1N);
    zero(out1, (long)NN * HO1N);
    zero(emax2, NN);             zero(den2, NN);
    zero(out2, (long)NN * HIDN);
    zero(pooled, (long)GGR * HIDN); zero(gcnt, GGR);

    // ---- weight prep: transpose to [Nout,K] f16 for WMMA B operand ----
    auto tr = [&](const float* in, _Float16* out, int R, int C) {
        k_transpose_h<<<cdiv_h((long)R * C, 256), 256, 0, stream>>>(in, out, R, C);
    };
    tr(Wl1, wl1t, 64, 256);  tr(Wr1, wr1t, 64, 256);  tr(We1, we1t, 32, 256);
    tr(Wl2, wl2t, 256, 64);  tr(Wr2, wr2t, 256, 64);  tr(We2, we2t, 32, 64);
    tr(Wo,  wot,  64, 128);

    // ---- node init (fused f16 output) ----
    k_node_init<<<NN, 64, 0, stream>>>(x, Wc, bc, Wi, bi, h0h);

    // ---- edge embedding + self-loop mean attr ----
    k_edge_embed<<<cdiv_h((long)EE * HEN, 256), 256, 0, stream>>>(ea, We, be, ee2);
    k_loop_accum<<<cdiv_h((long)EE * HEN, 256), 256, 0, stream>>>(ee2, tgtp, cnt);
    k_loop_final<<<cdiv_h((long)NN * HEN, 256), 256, 0, stream>>>(ee2, cnt);
    k_cvt<<<cdiv_h((long)E2E * HEN, 256), 256, 0, stream>>>(ee2, ee2h, (long)E2E * HEN);

    // ======== GATv2 layer 1 (H=4, HOUT=256) ========
    gemm(h0h, wl1t, bl1, xl1, NN, HO1N, HIDN);
    gemm(h0h, wr1t, br1, xr1, NN, HO1N, HIDN);
    gemm(ee2h, we1t, nullptr, ep, E2E, HO1N, HEN);
    k_scores<HO1N, H1N><<<cdiv_h(E2E, 4), 128, 0, stream>>>(
        xl1, xr1, ep, srcp, tgtp, EE, E2E, att1, sc1, emax1);
    k_exp<<<cdiv_h((long)E2E * H1N, 256), 256, 0, stream>>>(sc1, emax1, den1,
                                                            tgtp, EE, E2E, H1N);
    k_scatter<HO1N, H1N><<<cdiv_h(E2E, 4), 128, 0, stream>>>(
        xl1, srcp, tgtp, EE, E2E, sc1, den1, out1);
    k_epilogue<<<cdiv_h((long)NN * HO1N, 256), 256, 0, stream>>>(
        out1, bias1, nullptr, h1h, HO1N, (long)NN * HO1N);

    // ======== GATv2 layer 2 (H=1, HOUT=64) ========
    gemm(h1h, wl2t, bl2, xl2, NN, HIDN, HO1N);
    gemm(h1h, wr2t, br2, xr2, NN, HIDN, HO1N);
    gemm(ee2h, we2t, nullptr, ep, E2E, HIDN, HEN);
    k_scores<HIDN, 1><<<cdiv_h(E2E, 4), 128, 0, stream>>>(
        xl2, xr2, ep, srcp, tgtp, EE, E2E, att2, sc2, emax2);
    k_exp<<<cdiv_h(E2E, 256), 256, 0, stream>>>(sc2, emax2, den2, tgtp, EE, E2E, 1);
    k_scatter<HIDN, 1><<<cdiv_h(E2E, 4), 128, 0, stream>>>(
        xl2, srcp, tgtp, EE, E2E, sc2, den2, out2);
    k_epilogue<<<cdiv_h((long)NN * HIDN, 256), 256, 0, stream>>>(
        out2, bias2, out2, nullptr, HIDN, (long)NN * HIDN);

    // ======== mean pool + output projection ========
    k_pool<<<cdiv_h((long)NN * HIDN, 256), 256, 0, stream>>>(out2, batch, pooled, gcnt);
    k_pool_div<<<cdiv_h(GGR * HIDN, 256), 256, 0, stream>>>(pooled, gcnt, pooledh);
    gemm(pooledh, wot, bo, (float*)d_out, GGR, OUTN, HIDN);
}
